// CoAttention_47897475285202
// MI455X (gfx1250) — matrix operand order repaired and checked by reading, not compiled, not run on hardware
//
#include <hip/hip_runtime.h>
#include <hip/hip_bf16.h>
#include <cstdint>

#define BB 8
#define NN 2048
#define DD 128

typedef __attribute__((ext_vector_type(16))) _Float16 v16h;
typedef __attribute__((ext_vector_type(8)))  _Float16 v8h;
typedef __attribute__((ext_vector_type(8)))  float    v8f;

// ---------------- WMMA fragment helpers (wave32, CDNA5 layouts) ----------------

// 16x32 f16 A-fragment from row-major f16 source (ld in elements, multiple of 8).
// lane m = lane&15; lanes 0-15 hold K=[0..8)+[16..24), lanes 16-31 K=[8..16)+[24..32).
__device__ __forceinline__ v16h a_frag_f16(const _Float16* src, int ld) {
    const int lane  = threadIdx.x & 31;
    const int m     = lane & 15;
    const int khalf = (lane >> 4) << 3;
    const _Float16* p = src + (size_t)m * ld + khalf;
    v8h lo = *(const v8h*)(p);
    v8h hi = *(const v8h*)(p + 16);
    v16h a;
#pragma unroll
    for (int i = 0; i < 8; ++i) { a[i] = lo[i]; a[8 + i] = hi[i]; }
    return a;
}

// Same, converting from f32 on the fly.
__device__ __forceinline__ v16h a_frag_f32(const float* src, int ld) {
    const int lane  = threadIdx.x & 31;
    const int m     = lane & 15;
    const int khalf = (lane >> 4) << 3;
    const float* p = src + (size_t)m * ld + khalf;
    v16h a;
#pragma unroll
    for (int i = 0; i < 8; ++i) { a[i] = (_Float16)p[i]; a[8 + i] = (_Float16)p[16 + i]; }
    return a;
}

// 32x16 f16 B-fragment from row-major f16 [K x N]: lane n = lane&15; lanes 0-15 hold K=[0..8)+[16..24), lanes 16-31 K=[8..16)+[24..32).
__device__ __forceinline__ v16h b_frag_f16(const _Float16* src, size_t ld) {
    const int lane = threadIdx.x & 31;
    const _Float16* p = src + (size_t)((lane >> 4) << 3) * ld + (lane & 15);
    v16h b;
#pragma unroll
    for (int i = 0; i < 8; ++i) { b[i] = p[(size_t)i * ld]; b[8 + i] = p[(size_t)(16 + i) * ld]; }
    return b;
}

__device__ __forceinline__ v8f zero8() {
    v8f z;
#pragma unroll
    for (int i = 0; i < 8; ++i) z[i] = 0.0f;
    return z;
}

#define WMMA_F16(a, b, c) \
    __builtin_amdgcn_wmma_f32_16x16x32_f16(false, (a), false, (b), (short)0, (c), false, false)

__device__ __forceinline__ float redmax16(float v) {
#pragma unroll
    for (int m = 1; m <= 8; m <<= 1) v = fmaxf(v, __shfl_xor(v, m, 32));
    return v;
}
__device__ __forceinline__ float redsum16(float v) {
#pragma unroll
    for (int m = 1; m <= 8; m <<= 1) v += __shfl_xor(v, m, 32);
    return v;
}

// ---------------- Stage 0: transpose+convert the three weight matrices --------

__global__ __launch_bounds__(256) void prep_kernel(
    const float* __restrict__ Wk, const float* __restrict__ Wv, const float* __restrict__ Wo,
    _Float16* __restrict__ WkT, _Float16* __restrict__ WvT, _Float16* __restrict__ WoT) {
    const float* W = (blockIdx.x == 0) ? Wk : (blockIdx.x == 1) ? Wv : Wo;
    _Float16* T    = (blockIdx.x == 0) ? WkT : (blockIdx.x == 1) ? WvT : WoT;
    for (int i = threadIdx.x; i < DD * DD; i += blockDim.x) {
        int o = i >> 7, k = i & 127;
        T[k * DD + o] = (_Float16)W[i];   // T[d][out] = W[out][d]
    }
}

// ---------------- Stage 1: h = X@Wk^T, v = X@Wv^T (f16, + transposed h) -------

__global__ __launch_bounds__(128) void proj_kernel(
    const float* __restrict__ x1, const float* __restrict__ x2,
    const _Float16* __restrict__ WkT, const _Float16* __restrict__ WvT,
    _Float16* __restrict__ h1, _Float16* __restrict__ h2,
    _Float16* __restrict__ h1T, _Float16* __restrict__ h2T,
    _Float16* __restrict__ v1, _Float16* __restrict__ v2) {
    const int b = blockIdx.y;
    const int which = blockIdx.z;
    const float* X  = which ? x2  : x1;
    _Float16*    H  = which ? h2  : h1;
    _Float16*    HT = which ? h2T : h1T;
    _Float16*    V  = which ? v2  : v1;

    const int w = threadIdx.x >> 5, lane = threadIdx.x & 31;
    const int row0 = blockIdx.x * 64 + w * 16;
    const int n = lane & 15, mbase = (lane >> 4) * 8;

    const float* Xt = X + ((size_t)b * NN + row0) * DD;
    v16h a[4];
#pragma unroll
    for (int s = 0; s < 4; ++s) a[s] = a_frag_f32(Xt + s * 32, DD);

#pragma unroll
    for (int jo = 0; jo < 8; ++jo) {
        v8f ch = zero8(), cv = zero8();
#pragma unroll
        for (int s = 0; s < 4; ++s) {
            v16h bk = b_frag_f16(WkT + (size_t)(s * 32) * DD + jo * 16, DD);
            ch = WMMA_F16(a[s], bk, ch);
            v16h bv = b_frag_f16(WvT + (size_t)(s * 32) * DD + jo * 16, DD);
            cv = WMMA_F16(a[s], bv, cv);
        }
        v8h pack;
#pragma unroll
        for (int r = 0; r < 8; ++r) {
            _Float16 e = (_Float16)ch[r];
            size_t ridx = ((size_t)b * NN + row0 + mbase + r) * DD + jo * 16 + n;
            H[ridx] = e;
            V[ridx] = (_Float16)cv[r];
            pack[r] = e;
        }
        *(v8h*)(HT + ((size_t)b * DD + jo * 16 + n) * NN + row0 + mbase) = pack;
    }
}

// ------- Stage 2: online row-softmax stats of E = hA @ hBT (both dirs) --------
// LDS-staged, double-buffered 64-column chunks; gridDim.z selects direction.

__global__ __launch_bounds__(128) void stats_kernel(
    const _Float16* __restrict__ h1, const _Float16* __restrict__ h2T,
    const _Float16* __restrict__ h2, const _Float16* __restrict__ h1T,
    float* __restrict__ m12, float* __restrict__ s12,
    float* __restrict__ m21, float* __restrict__ s21) {
    __shared__ __align__(32) _Float16 ldsB[2][128 * 72];   // 64-col chunk, rows padded to 72

    const int dir = blockIdx.z;
    const _Float16* hA  = dir ? h2  : h1;
    const _Float16* hBT = dir ? h1T : h2T;
    float* omax = dir ? m21 : m12;
    float* osum = dir ? s21 : s12;

    const int b = blockIdx.y;
    const int tid = threadIdx.x;
    const int w = tid >> 5, lane = tid & 31;
    const int row0 = blockIdx.x * 64 + w * 16;
    const int n = lane & 15, mbase = (lane >> 4) * 8;

    const _Float16* Abase = hA + ((size_t)b * NN + row0) * DD;
    v16h a[4];
#pragma unroll
    for (int s = 0; s < 4; ++s) a[s] = a_frag_f16(Abase + s * 32, DD);

    float rmax[8], rsum[8];
#pragma unroll
    for (int r = 0; r < 8; ++r) { rmax[r] = -3.0e38f; rsum[r] = 0.0f; }

    const _Float16* Bbase = hBT + (size_t)b * DD * NN;

    // preload chunk 0: 128 rows x 64 halves  (8 x v8h per thread)
#pragma unroll
    for (int i = 0; i < 8; ++i) {
        int flat = i * 128 + tid;
        int row = flat >> 3, seg = flat & 7;
        *(v8h*)&ldsB[0][row * 72 + seg * 8] =
            *(const v8h*)(Bbase + (size_t)row * NN + seg * 8);
    }
    __syncthreads();

    const int NCHUNK = NN / 64;
    for (int c = 0; c < NCHUNK; ++c) {
        const int cur = c & 1;
        const bool have = (c + 1) < NCHUNK;
        v8h nx[8];
        if (have) {
            const size_t jn = (size_t)(c + 1) * 64;
#pragma unroll
            for (int i = 0; i < 8; ++i) {
                int flat = i * 128 + tid;
                int row = flat >> 3, seg = flat & 7;
                nx[i] = *(const v8h*)(Bbase + (size_t)row * NN + jn + seg * 8);
            }
        }
#pragma unroll
        for (int jt = 0; jt < 4; ++jt) {
            v8f cc = zero8();
#pragma unroll
            for (int s = 0; s < 4; ++s) {
                v16h bb = b_frag_f16(&ldsB[cur][(s * 32) * 72 + jt * 16], 72);
                cc = WMMA_F16(a[s], bb, cc);
            }
#pragma unroll
            for (int r = 0; r < 8; ++r) {
                float v    = cc[r];
                float tmax = redmax16(v);
                float nm   = fmaxf(rmax[r], tmax);
                float te   = redsum16(__expf(v - nm));
                rsum[r] = rsum[r] * __expf(rmax[r] - nm) + te;
                rmax[r] = nm;
            }
        }
        if (have) {
#pragma unroll
            for (int i = 0; i < 8; ++i) {
                int flat = i * 128 + tid;
                int row = flat >> 3, seg = flat & 7;
                *(v8h*)&ldsB[1 - cur][row * 72 + seg * 8] = nx[i];
            }
        }
        __syncthreads();
    }

    if (n == 0) {
#pragma unroll
        for (int r = 0; r < 8; ++r) {
            omax[(size_t)b * NN + row0 + mbase + r] = rmax[r];
            osum[(size_t)b * NN + row0 + mbase + r] = rsum[r];
        }
    }
}

// ---- Stage 3: out = leakyrelu((softmax(E)@V)@Wo^T + bo), fused, both dirs ----
// LDS-staged, double-buffered 32-column chunks of hBT and V.

#define B_BUF   (128 * 40)        // one B chunk (rows padded to 40 halves)
#define V_BUF   (32 * 136)        // one V chunk (rows padded to 136 halves)
#define V_OFF   (2 * B_BUF)
#define P_OFF   (2 * B_BUF + 2 * V_BUF)

__global__ __launch_bounds__(128) void attend_kernel(
    const _Float16* __restrict__ h1, const _Float16* __restrict__ h2T,
    const _Float16* __restrict__ v2f,
    const float* __restrict__ m21, const float* __restrict__ s21,
    const _Float16* __restrict__ h2, const _Float16* __restrict__ h1T,
    const _Float16* __restrict__ v1f,
    const float* __restrict__ m12, const float* __restrict__ s12,
    const _Float16* __restrict__ WoT, const float* __restrict__ bo,
    float* __restrict__ out1, float* __restrict__ out2) {
    // one shared block, carved by offsets (no pointer arrays of LDS addresses);
    // the epilogue n-tile region aliases the B-chunk region behind a barrier.
    __shared__ __align__(32) _Float16 smem[P_OFF + 4 * 16 * 40];

    const int dir = blockIdx.z;
    const _Float16* hA   = dir ? h2  : h1;
    const _Float16* hBT  = dir ? h1T : h2T;
    const _Float16* V    = dir ? v1f : v2f;
    const float*    cmax = dir ? m12 : m21;
    const float*    csum = dir ? s12 : s21;
    float*          out  = dir ? out2 : out1;

    const int b = blockIdx.y;
    const int tid = threadIdx.x;
    const int w = tid >> 5, lane = tid & 31;
    const int row0 = blockIdx.x * 64 + w * 16;
    const int n = lane & 15, mbase = (lane >> 4) * 8;

    const _Float16* Abase = hA + ((size_t)b * NN + row0) * DD;
    v16h a[4];
#pragma unroll
    for (int s = 0; s < 4; ++s) a[s] = a_frag_f16(Abase + s * 32, DD);

    const _Float16* Bbase = hBT + (size_t)b * DD * NN;
    const _Float16* Vbase = V   + (size_t)b * NN * DD;

    v8f acc[8];
#pragma unroll
    for (int i = 0; i < 8; ++i) acc[i] = zero8();

    // preload chunk 0: B = 128 rows x 32 halves, V = 32 rows x 128 halves
#pragma unroll
    for (int i = 0; i < 4; ++i) {
        int flat = i * 128 + tid;
        int row = flat >> 2, seg = flat & 3;
        *(v8h*)&smem[row * 40 + seg * 8] =
            *(const v8h*)(Bbase + (size_t)row * NN + seg * 8);
        int vrow = flat >> 4, vseg = flat & 15;
        *(v8h*)&smem[V_OFF + vrow * 136 + vseg * 8] =
            *(const v8h*)(Vbase + (size_t)vrow * DD + vseg * 8);
    }
    __syncthreads();

    const int NCHUNK = NN / 32;
    for (int c = 0; c < NCHUNK; ++c) {
        const int cur = c & 1;
        const int j0 = c * 32;
        const bool have = (c + 1) < NCHUNK;
        _Float16* bcur = smem + cur * B_BUF;
        _Float16* vcur = smem + V_OFF + cur * V_BUF;
        v8h nB[4], nV[4];
        if (have) {
            const size_t jn = (size_t)j0 + 32;
#pragma unroll
            for (int i = 0; i < 4; ++i) {
                int flat = i * 128 + tid;
                int row = flat >> 2, seg = flat & 3;
                nB[i] = *(const v8h*)(Bbase + (size_t)row * NN + jn + seg * 8);
                int vrow = flat >> 4, vseg = flat & 15;
                nV[i] = *(const v8h*)(Vbase + (jn + vrow) * DD + vseg * 8);
            }
        }
        // two 16-col E tiles -> column-normalized P into per-wave LDS
        _Float16* ldsP = smem + P_OFF + w * (16 * 40);
#pragma unroll
        for (int t = 0; t < 2; ++t) {
            const int j = j0 + t * 16;
            v8f cc = zero8();
#pragma unroll
            for (int s = 0; s < 4; ++s) {
                v16h bb = b_frag_f16(bcur + (s * 32) * 40 + t * 16, 40);
                cc = WMMA_F16(a[s], bb, cc);
            }
            float cm = cmax[(size_t)b * NN + j + n];
            float cr = 1.0f / csum[(size_t)b * NN + j + n];
#pragma unroll
            for (int r = 0; r < 8; ++r) {
                float p = __expf(cc[r] - cm) * cr;
                ldsP[(mbase + r) * 40 + t * 16 + n] = (_Float16)p;
            }
        }
        // D-layout -> A-layout (intra-wave), then P @ V
        v16h pf = a_frag_f16(ldsP, 40);
#pragma unroll
        for (int jd = 0; jd < 8; ++jd) {
            v16h bv = b_frag_f16(vcur + jd * 16, 136);
            acc[jd] = WMMA_F16(pf, bv, acc[jd]);
        }
        if (have) {
            _Float16* bnxt = smem + (1 - cur) * B_BUF;
            _Float16* vnxt = smem + V_OFF + (1 - cur) * V_BUF;
#pragma unroll
            for (int i = 0; i < 4; ++i) {
                int flat = i * 128 + tid;
                int row = flat >> 2, seg = flat & 3;
                *(v8h*)&bnxt[row * 40 + seg * 8] = nB[i];
                int vrow = flat >> 4, vseg = flat & 15;
                *(v8h*)&vnxt[vrow * 136 + vseg * 8] = nV[i];
            }
        }
        __syncthreads();
    }

    __syncthreads();   // retire all chunk-buffer readers before aliasing as ldsN

    // Fused epilogue: (n-tile) @ Wo^T + bo, leaky-ReLU(0.01)
    _Float16* ldsN = smem + w * (16 * 136);   // aliases B-chunk region (safe post-barrier)
#pragma unroll
    for (int jd = 0; jd < 8; ++jd)
#pragma unroll
        for (int r = 0; r < 8; ++r)
            ldsN[(mbase + r) * 136 + jd * 16 + n] = (_Float16)acc[jd][r];

    v16h na[4];
#pragma unroll
    for (int s = 0; s < 4; ++s) na[s] = a_frag_f16(ldsN + s * 32, 136);

#pragma unroll
    for (int jo = 0; jo < 8; ++jo) {
        v8f m = zero8();
#pragma unroll
        for (int s = 0; s < 4; ++s) {
            v16h bw = b_frag_f16(WoT + (size_t)(s * 32) * DD + jo * 16, DD);
            m = WMMA_F16(na[s], bw, m);
        }
        float bias = bo[jo * 16 + n];
#pragma unroll
        for (int r = 0; r < 8; ++r) {
            float v = m[r] + bias;
            v = (v > 0.0f) ? v : 0.01f * v;
            out[((size_t)b * NN + row0 + mbase + r) * DD + jo * 16 + n] = v;
        }
    }
}

// ---------------- Host-side orchestration ------------------------------------

extern "C" void kernel_launch(void* const* d_in, const int* in_sizes, int n_in,
                              void* d_out, int out_size, void* d_ws, size_t ws_size,
                              hipStream_t stream) {
    const float* x1 = (const float*)d_in[0];
    const float* x2 = (const float*)d_in[1];
    const float* Wk = (const float*)d_in[2];
    const float* Wv = (const float*)d_in[3];
    const float* Wo = (const float*)d_in[4];
    const float* bo = (const float*)d_in[5];

    char* ws = (char*)d_ws;
    size_t off = 0;
    auto take = [&](size_t bytes) -> char* {
        char* p = ws + off;
        off += (bytes + 255) & ~(size_t)255;
        return p;
    };
    const size_t WT_BYTES = (size_t)DD * DD * sizeof(_Float16);
    const size_t HV_BYTES = (size_t)BB * NN * DD * sizeof(_Float16);
    const size_t ST_BYTES = (size_t)BB * NN * sizeof(float);

    _Float16* WkT = (_Float16*)take(WT_BYTES);
    _Float16* WvT = (_Float16*)take(WT_BYTES);
    _Float16* WoT = (_Float16*)take(WT_BYTES);
    _Float16* h1  = (_Float16*)take(HV_BYTES);
    _Float16* h2  = (_Float16*)take(HV_BYTES);
    _Float16* h1T = (_Float16*)take(HV_BYTES);
    _Float16* h2T = (_Float16*)take(HV_BYTES);
    _Float16* v1  = (_Float16*)take(HV_BYTES);
    _Float16* v2  = (_Float16*)take(HV_BYTES);
    float* r12max = (float*)take(ST_BYTES);   // row stats of e12
    float* r12sum = (float*)take(ST_BYTES);
    float* r21max = (float*)take(ST_BYTES);   // row stats of e21 == column stats of e12
    float* r21sum = (float*)take(ST_BYTES);

    prep_kernel<<<3, 256, 0, stream>>>(Wk, Wv, Wo, WkT, WvT, WoT);

    proj_kernel<<<dim3(NN / 64, BB, 2), 128, 0, stream>>>(
        x1, x2, WkT, WvT, h1, h2, h1T, h2T, v1, v2);

    stats_kernel<<<dim3(NN / 64, BB, 2), 128, 0, stream>>>(
        h1, h2T, h2, h1T, r12max, r12sum, r21max, r21sum);

    float* out1 = (float*)d_out;
    float* out2 = out1 + (size_t)BB * NN * DD;
    attend_kernel<<<dim3(NN / 64, BB, 2), 128, 0, stream>>>(
        h1, h2T, v2, r21max, r21sum,
        h2, h1T, v1, r12max, r12sum,
        WoT, bo, out1, out2);
}
